// RCCA_57011395887300
// MI455X (gfx1250) — matrix operand order repaired and checked by reading, not compiled, not run on hardware
//
#include <hip/hip_runtime.h>
#include <stdint.h>

// ---------------------------------------------------------------------------
// Types for CDNA5 WMMA (gfx1250, wave32)
// ---------------------------------------------------------------------------
typedef __attribute__((ext_vector_type(16))) __bf16 v16bf;
typedef __attribute__((ext_vector_type(8)))  __bf16 v8bf;
typedef __attribute__((ext_vector_type(8)))  float  v8f;

__device__ __forceinline__ __bf16 f2bf(float f) {
  union { float f; uint32_t u; } v; v.f = f;
  uint32_t u = v.u + 0x7FFFu + ((v.u >> 16) & 1u);   // round-to-nearest-even
  union { uint16_t s; __bf16 b; } o; o.s = (uint16_t)(u >> 16);
  return o.b;
}
__device__ __forceinline__ float bf2f(__bf16 b) {
  union { uint16_t s; __bf16 b; } i; i.b = b;
  union { uint32_t u; float f; } o; o.u = ((uint32_t)i.s) << 16;
  return o.f;
}

// Problem constants
#define PH    97
#define PW    97
#define PHW   9409          // 97*97
#define NTOT  18818         // 2*97*97
#define WP    99
#define HPWP  9801          // 99*99
#define NTILES 295          // ceil(18818/64)

// ---------------------------------------------------------------------------
// BN fold: s = g*rsqrt(v+eps), t = b - m*s
// ---------------------------------------------------------------------------
__global__ void bn_prep(const float* __restrict__ g, const float* __restrict__ b,
                        const float* __restrict__ m, const float* __restrict__ v,
                        float* __restrict__ s, float* __restrict__ t, int C) {
  int c = blockIdx.x * 256 + threadIdx.x;
  if (c < C) {
    float sc = g[c] * rsqrtf(v[c] + 1e-5f);
    s[c] = sc;
    t[c] = b[c] - m[c] * sc;
  }
}

// ---------------------------------------------------------------------------
// Pack OIHW fp32 weights into fragment-ready bf16 A tiles.
// K is REORDERED as K' = (kh*3+kw)*Cin + ci  (channel-minor) so that a 32-deep
// K step in the GEMM stays inside a single (kh,kw) plane (Cin % 32 == 0).
// A fragment (16x32 bf16) lane layout per ISA:
//   lane L: row M = L%16 ; base = (L>=16)*8
//   halves t in [0,8):  K = base + t          (VGPR 0..3)
//   halves t in [8,16): K = 16 + base + (t-8) (VGPR 4..7)
// Storage: apack[fragIdx][lane][16 halves], fragIdx = mo*(Ktot/32)+ko
// M is padded to Mpad (multiple of 128); rows >= Cout pack zeros.
// ---------------------------------------------------------------------------
__global__ void pack_wfrag(const float* __restrict__ w, __bf16* __restrict__ apack,
                           int Cout, int Cin, int ks3, int Ktot, int total) {
  int e = blockIdx.x * 256 + threadIdx.x;
  if (e >= total) return;
  int f = e >> 9;            // / 512
  int r = e & 511;
  int lane = r >> 4;
  int t = r & 15;
  int nK = Ktot >> 5;
  int mo = f / nK;
  int ko = f - mo * nK;
  int M = mo * 16 + (lane & 15);
  int base = (lane >> 4) * 8;
  int K = (ko << 5) + ((t < 8) ? (base + t) : (16 + base + (t - 8)));
  float val = 0.f;
  if (M < Cout) {
    if (ks3) {
      int rr = K / Cin;              // kh*3+kw
      int ci = K - rr * Cin;
      int kh = rr / 3, kw = rr - kh * 3;
      val = w[(((size_t)M * Cin + ci) * 3 + kh) * 3 + kw];
    } else {
      val = w[(size_t)M * Cin + K];
    }
  }
  apack[e] = f2bf(val);
}

// ---------------------------------------------------------------------------
// Fill interior of a zero-padded bf16 image (99x99) from unpadded src
// ---------------------------------------------------------------------------
__global__ void pad_fill_f32(const float* __restrict__ src, __bf16* __restrict__ dst,
                             int C, int dst_batch_stride, int total) {
  int i = blockIdx.x * 256 + threadIdx.x;
  if (i >= total) return;
  int b = i / (C * PHW);
  int r = i - b * (C * PHW);
  int c = r / PHW;
  int rem = r - c * PHW;
  int h = rem / PW, w = rem - h * PW;
  dst[(size_t)b * dst_batch_stride + (size_t)c * HPWP + (h + 1) * WP + (w + 1)] = f2bf(src[i]);
}

__global__ void pad_fill_bf16(const __bf16* __restrict__ src, __bf16* __restrict__ dst,
                              int C, int dst_batch_stride, int total) {
  int i = blockIdx.x * 256 + threadIdx.x;
  if (i >= total) return;
  int b = i / (C * PHW);
  int r = i - b * (C * PHW);
  int c = r / PHW;
  int rem = r - c * PHW;
  int h = rem / PW, w = rem - h * PW;
  dst[(size_t)b * dst_batch_stride + (size_t)c * HPWP + (h + 1) * WP + (w + 1)] = src[i];
}

// ---------------------------------------------------------------------------
// Implicit-GEMM convolution with v_wmma_f32_16x16x32_bf16.
//   M = Cout (padded to mult of 128), N = B*H*W = 18818, K = ks*ks*Cin
// Block: 256 threads (8 wave32), tile 128(M) x 64(N), K step 32.
// Each wave owns a 2x2 grid of C fragments -> 4 WMMAs per K step.
// A fragments come straight from pre-packed global (2x b128 per frag/lane);
// B tile (32K x 64N) is gathered into double-buffered LDS. K is ordered
// (kh,kw)-major / channel-minor and CH (input channel stride) is a template
// literal, so each thread's 8 gathers are immediate-offset loads off a single
// 32-bit voffset and land in LDS via ONE ds_store_b128.
// Epilogue: acc*scale + bias (+ReLU), fp32 or bf16 store, optionally into a
// padded layout (fuses convb output into the concat buffer for bott).
// ---------------------------------------------------------------------------
#define BSS 40   // LDS row stride in halves for the 64x32 B tile (bank-friendly)

template <int KS3, int CH>
__global__ __launch_bounds__(256) void wmma_conv_gemm(
    const __bf16* __restrict__ in, const __bf16* __restrict__ apack,
    const float* __restrict__ scale, const float* __restrict__ bias,
    float* __restrict__ outf, __bf16* __restrict__ outb,
    int in_batch_stride,
    int out_ch_stride, int out_row_stride, int out_batch_stride,
    int out_pad, int relu, int Cin, int M) {
  __shared__ __bf16 Bs[2][64 * BSS];
  __shared__ int base_in[64];
  __shared__ int base_out[64];

  const int tid = threadIdx.x;

  if (tid < 64) {
    int n = blockIdx.x * 64 + tid;
    if (n < NTOT) {
      int b = n / PHW;
      int rem = n - b * PHW;
      int h = rem / PW, w = rem - h * PW;
      base_in[tid]  = b * in_batch_stride + (KS3 ? (h * WP + w) : rem);
      base_out[tid] = b * out_batch_stride +
                      (out_pad ? ((h + 1) * out_row_stride + (w + 1)) : rem);
    } else {
      base_in[tid] = -1;
      base_out[tid] = -1;
    }
  }
  __syncthreads();

  const int lane = tid & 31;
  const int wid  = tid >> 5;
  const int mg   = wid & 3;   // m-fragment pair: frags 2*mg, 2*mg+1 (of 8)
  const int ng   = wid >> 2;  // n-fragment pair: frags 2*ng, 2*ng+1 (of 4)

  const int nK    = (KS3 ? 9 : 1) * (Cin >> 5);
  const int moG0  = blockIdx.y * 8 + mg * 2;
  const __bf16* afp0 = apack + (size_t)moG0 * nK * 512 + (size_t)lane * 16;
  const __bf16* afp1 = afp0 + (size_t)nK * 512;

  // B-tile fill: this thread owns column nn, K rows [8*kkb, 8*kkb+8)
  const int nn   = tid & 63;
  const int kkb  = tid >> 6;
  const int bi   = base_in[nn];
  const int kcs0 = (kkb * 8) * CH;                       // element offset of first owned K row
  __bf16* bsl0 = &Bs[0][nn * BSS + kkb * 8];             // 16B-aligned
  __bf16* bsl1 = &Bs[1][nn * BSS + kkb * 8];

  // OOB columns: zero both buffers' slots once; never rewritten.
  if (bi < 0) {
    v8bf z = {};
    *(v8bf*)bsl0 = z;
    *(v8bf*)bsl1 = z;
  }

  // B-fragment LDS read offsets (B 32x16: col = lane%16, K = (lane/16)*8 + [0,8) and 16 + (lane/16)*8 + [0,8))
  const int bc0 = (ng * 32 + (lane & 15)) * BSS + ((lane >> 4) * 8);
  const int bc1 = bc0 + 16 * BSS;

  v8f acc[4] = {v8f{}, v8f{}, v8f{}, v8f{}};

  // Uniform scalar gather trackers (no divisions in the hot loop)
  int offF = 0;                       // spatial offset of (kh,kw) being FILLED
  int cbF  = 0;                       // channel-block element offset being FILLED
  int kwF  = 0;
  const int cbWrap = Cin * CH;

  auto fill = [&](__bf16* dst, int offuni) {
    if (bi >= 0) {
      int idx = bi + kcs0 + offuni;    // 32-bit voffset; base stays in SGPRs
      v8bf vv;
#pragma unroll
      for (int j = 0; j < 8; ++j) vv[j] = in[idx + j * CH];  // imm-offset loads
      *(v8bf*)dst = vv;                // single ds_store_b128
    }
  };
  auto advance = [&]() {
    cbF += 32 * CH;
    if (cbF == cbWrap) {
      cbF = 0;
      if (KS3) {
        if (kwF == 2) { kwF = 0; offF += WP - 2; }
        else          { kwF++;   offF += 1; }
      }
    }
  };

  fill(bsl0, offF + cbF);
  advance();

  for (int ko = 0; ko < nK; ++ko) {
    __syncthreads();                      // fill(ko) complete across block
    if (ko + 1 < nK) {                    // overlap next gather with WMMAs
      fill((ko & 1) ? bsl0 : bsl1, offF + cbF);
      advance();
      __builtin_prefetch(afp0 + (size_t)(ko + 1) * 512, 0, 1);
      __builtin_prefetch(afp1 + (size_t)(ko + 1) * 512, 0, 1);
    }

    const __bf16* bs = Bs[ko & 1];
    v16bf a0 = *(const v16bf*)(afp0 + (size_t)ko * 512);
    v16bf a1 = *(const v16bf*)(afp1 + (size_t)ko * 512);

    v8bf b0lo = *(const v8bf*)(bs + bc0);
    v8bf b0hi = *(const v8bf*)(bs + bc0 + 16);
    v8bf b1lo = *(const v8bf*)(bs + bc1);
    v8bf b1hi = *(const v8bf*)(bs + bc1 + 16);
    v16bf b0 = __builtin_shufflevector(b0lo, b0hi, 0,1,2,3,4,5,6,7,8,9,10,11,12,13,14,15);
    v16bf b1 = __builtin_shufflevector(b1lo, b1hi, 0,1,2,3,4,5,6,7,8,9,10,11,12,13,14,15);

    acc[0] = __builtin_amdgcn_wmma_f32_16x16x32_bf16(false, a0, false, b0, (short)0, acc[0], false, false);
    acc[1] = __builtin_amdgcn_wmma_f32_16x16x32_bf16(false, a0, false, b1, (short)0, acc[1], false, false);
    acc[2] = __builtin_amdgcn_wmma_f32_16x16x32_bf16(false, a1, false, b0, (short)0, acc[2], false, false);
    acc[3] = __builtin_amdgcn_wmma_f32_16x16x32_bf16(false, a1, false, b1, (short)0, acc[3], false, false);
  }

  // Epilogue: C 16x16 layout => VGPR r: lanes 0-15 -> M=r, lanes 16-31 -> M=8+r
#pragma unroll
  for (int am = 0; am < 2; ++am) {
#pragma unroll
    for (int r = 0; r < 8; ++r) {
      int m = (lane < 16) ? r : (r + 8);
      int co = blockIdx.y * 128 + (mg * 2 + am) * 16 + m;
      if (co < M) {
        float s  = scale ? scale[co] : 1.0f;
        float tt = bias ? bias[co] : 0.0f;
#pragma unroll
        for (int p = 0; p < 2; ++p) {
          int nl = (ng * 2 + p) * 16 + (lane & 15);
          int bo = base_out[nl];
          if (bo >= 0) {
            float vv = acc[am * 2 + p][r] * s + tt;
            if (relu) vv = fmaxf(vv, 0.f);
            size_t o = (size_t)bo + (size_t)co * out_ch_stride;
            if (outf) outf[o] = vv;
            else      outb[o] = f2bf(vv);
          }
        }
      }
    }
  }
}

// ---------------------------------------------------------------------------
// Criss-cross attention: one block per (b,h,w). Scores over the column (i) and
// row (j), joint softmax over 194 entries (self-column-position masked),
// aggregation over v, residual + gamma applied in place on `cur`.
// ---------------------------------------------------------------------------
__global__ __launch_bounds__(256) void cca_attention(
    const __bf16* __restrict__ q, const __bf16* __restrict__ k,
    const __bf16* __restrict__ v, __bf16* __restrict__ cur,
    const float* __restrict__ gamma) {
  const int Cq = 64, C = 512;
  int n = blockIdx.x;
  int b = n / PHW;
  int rem = n - b * PHW;
  int h = rem / PW, w = rem - h * PW;
  int t = threadIdx.x;

  __shared__ float qv[64];
  __shared__ float sc[194];
  __shared__ float red[2];

  if (t < 64) qv[t] = bf2f(q[((size_t)b * Cq + t) * PHW + rem]);
  __syncthreads();

  if (t < PH) {                       // column scores: i = t
    float s;
    if (t == h) {
      s = -INFINITY;                  // self mask on the column direction
    } else {
      s = 0.f;
      const __bf16* kp = k + (size_t)b * Cq * PHW + t * PW + w;
      for (int c = 0; c < 64; ++c) s += qv[c] * bf2f(kp[(size_t)c * PHW]);
    }
    sc[t] = s;
  } else if (t >= 128 && t < 128 + PW) {  // row scores: j = t-128
    int j = t - 128;
    float s = 0.f;
    const __bf16* kp = k + (size_t)b * Cq * PHW + h * PW + j;
    for (int c = 0; c < 64; ++c) s += qv[c] * bf2f(kp[(size_t)c * PHW]);
    sc[PH + j] = s;
  }
  __syncthreads();

  if (t == 0) {
    float m = -INFINITY;
    for (int i = 0; i < 194; ++i) m = fmaxf(m, sc[i]);
    red[0] = m;
  }
  __syncthreads();
  float m = red[0];
  for (int i = t; i < 194; i += 256) sc[i] = __expf(sc[i] - m);
  __syncthreads();
  if (t == 0) {
    float s = 0.f;
    for (int i = 0; i < 194; ++i) s += sc[i];
    red[1] = 1.0f / s;
  }
  __syncthreads();
  float inv = red[1];
  float g = gamma[0];

  for (int c = t; c < C; c += 256) {
    const __bf16* vc = v + ((size_t)b * C + c) * PHW;
    float acc = 0.f;
    for (int i = 0; i < PH; ++i) acc += sc[i] * bf2f(vc[i * PW + w]);
    const __bf16* vr = vc + h * PW;
    for (int j = 0; j < PW; ++j) acc += sc[PH + j] * bf2f(vr[j]);
    size_t o = ((size_t)b * C + c) * PHW + rem;
    cur[o] = f2bf(g * acc * inv + bf2f(cur[o]));
  }
}

// ---------------------------------------------------------------------------
// Host-side orchestration
// ---------------------------------------------------------------------------
static inline size_t alignup(size_t x) { return (x + 255) & ~(size_t)255; }

extern "C" void kernel_launch(void* const* d_in, const int* in_sizes, int n_in,
                              void* d_out, int out_size, void* d_ws, size_t ws_size,
                              hipStream_t stream) {
  (void)in_sizes; (void)n_in; (void)out_size; (void)ws_size;

  const float* x        = (const float*)d_in[0];
  const float* conva_w  = (const float*)d_in[1];
  const float* ca_g = (const float*)d_in[2];
  const float* ca_b = (const float*)d_in[3];
  const float* ca_m = (const float*)d_in[4];
  const float* ca_v = (const float*)d_in[5];
  const float* q_w  = (const float*)d_in[6];
  const float* q_b  = (const float*)d_in[7];
  const float* k_w  = (const float*)d_in[8];
  const float* k_b  = (const float*)d_in[9];
  const float* v_w  = (const float*)d_in[10];
  const float* v_b  = (const float*)d_in[11];
  const float* gamma = (const float*)d_in[12];
  const float* convb_w = (const float*)d_in[13];
  const float* cb_g = (const float*)d_in[14];
  const float* cb_b = (const float*)d_in[15];
  const float* cb_m = (const float*)d_in[16];
  const float* cb_v = (const float*)d_in[17];
  const float* bott_w = (const float*)d_in[18];
  const float* bt_g = (const float*)d_in[19];
  const float* bt_b = (const float*)d_in[20];
  const float* bt_m = (const float*)d_in[21];
  const float* bt_v = (const float*)d_in[22];
  // d_in[23] = recurrence (device int); setup fixes it to 2 and graph capture
  // forbids sync readback, so the two CCA rounds are unrolled below.

  float* outp = (float*)d_out;

  // --- workspace layout (bump allocator, 256B aligned) ---
  char* ws = (char*)d_ws;
  size_t off = 0;
  auto alloc = [&](size_t bytes) { char* p = ws + off; off += alignup(bytes); return p; };

  const size_t padBig_e = (size_t)2 * 2560 * HPWP;  // x(2048ch) ++ convb out(512ch), padded
  const size_t padB_e   = (size_t)2 * 512 * HPWP;   // attention output, padded
  const size_t a1_e     = (size_t)2 * 512 * PHW;
  const size_t qk_e     = (size_t)2 * 64 * PHW;

  __bf16* padBig = (__bf16*)alloc(padBig_e * 2);
  __bf16* padB   = (__bf16*)alloc(padB_e * 2);
  __bf16* a1     = (__bf16*)alloc(a1_e * 2);
  __bf16* qb     = (__bf16*)alloc(qk_e * 2);
  __bf16* kb     = (__bf16*)alloc(qk_e * 2);
  __bf16* vb     = (__bf16*)alloc(a1_e * 2);
  __bf16* apA    = (__bf16*)alloc((size_t)512 * 18432 * 2);
  __bf16* apB    = (__bf16*)alloc((size_t)512 * 4608 * 2);
  __bf16* apBot  = (__bf16*)alloc((size_t)512 * 23040 * 2);
  __bf16* apQ    = (__bf16*)alloc((size_t)128 * 512 * 2);   // M padded 64->128
  __bf16* apK    = (__bf16*)alloc((size_t)128 * 512 * 2);
  __bf16* apV    = (__bf16*)alloc((size_t)512 * 512 * 2);
  float*  sA = (float*)alloc(512 * 4);  float* tA = (float*)alloc(512 * 4);
  float*  sB = (float*)alloc(512 * 4);  float* tB = (float*)alloc(512 * 4);
  float*  sT = (float*)alloc(512 * 4);  float* tT = (float*)alloc(512 * 4);

  // --- zero padded buffers (borders must be 0 for the 3x3 implicit GEMM) ---
  hipMemsetAsync(padBig, 0, padBig_e * 2, stream);
  hipMemsetAsync(padB,   0, padB_e * 2,   stream);

  // --- fold BN params ---
  bn_prep<<<2, 256, 0, stream>>>(ca_g, ca_b, ca_m, ca_v, sA, tA, 512);
  bn_prep<<<2, 256, 0, stream>>>(cb_g, cb_b, cb_m, cb_v, sB, tB, 512);
  bn_prep<<<2, 256, 0, stream>>>(bt_g, bt_b, bt_m, bt_v, sT, tT, 512);

  // --- pack weights into WMMA A fragments (K reordered, M padded) ---
  auto pack = [&](const float* w, __bf16* ap, int Cout, int Cin, int ks3,
                  int Ktot, int Mpad) {
    int total = Mpad * Ktot;
    pack_wfrag<<<(total + 255) / 256, 256, 0, stream>>>(w, ap, Cout, Cin, ks3, Ktot, total);
  };
  pack(conva_w, apA,   512, 2048, 1, 18432, 512);
  pack(convb_w, apB,   512,  512, 1,  4608, 512);
  pack(bott_w,  apBot, 512, 2560, 1, 23040, 512);
  pack(q_w,     apQ,    64,  512, 0,   512, 128);
  pack(k_w,     apK,    64,  512, 0,   512, 128);
  pack(v_w,     apV,   512,  512, 0,   512, 512);

  // --- x -> padded bf16 (channels 0..2047 of the concat buffer) ---
  {
    int total = 2 * 2048 * PHW;
    pad_fill_f32<<<(total + 255) / 256, 256, 0, stream>>>(x, padBig, 2048, 2560 * HPWP, total);
  }

  // --- conva: 3x3 2048->512, BN+ReLU, out a1 (bf16) ---
  wmma_conv_gemm<1, HPWP><<<dim3(NTILES, 4), 256, 0, stream>>>(
      padBig, apA, sA, tA, nullptr, a1,
      2560 * HPWP,
      PHW, PW, 512 * PHW, /*out_pad*/0, /*relu*/1, 2048, 512);

  // --- 2 rounds of criss-cross attention (recurrence fixed at 2) ---
  for (int r = 0; r < 2; ++r) {
    wmma_conv_gemm<0, PHW><<<dim3(NTILES, 1), 256, 0, stream>>>(
        a1, apQ, nullptr, q_b, nullptr, qb,
        512 * PHW, PHW, PW, 64 * PHW, 0, 0, 512, 64);
    wmma_conv_gemm<0, PHW><<<dim3(NTILES, 1), 256, 0, stream>>>(
        a1, apK, nullptr, k_b, nullptr, kb,
        512 * PHW, PHW, PW, 64 * PHW, 0, 0, 512, 64);
    wmma_conv_gemm<0, PHW><<<dim3(NTILES, 4), 256, 0, stream>>>(
        a1, apV, nullptr, v_b, nullptr, vb,
        512 * PHW, PHW, PW, 512 * PHW, 0, 0, 512, 512);
    cca_attention<<<NTOT, 256, 0, stream>>>(qb, kb, vb, a1, gamma);
  }

  // --- a1 -> padded bf16 for convb ---
  {
    int total = 2 * 512 * PHW;
    pad_fill_bf16<<<(total + 255) / 256, 256, 0, stream>>>(a1, padB, 512, 512 * HPWP, total);
  }

  // --- convb: 3x3 512->512, BN+ReLU; store directly into the padded concat
  //     buffer at channel offset 2048 (fuses the concat with x) ---
  wmma_conv_gemm<1, HPWP><<<dim3(NTILES, 4), 256, 0, stream>>>(
      padB, apB, sB, tB, nullptr, padBig + (size_t)2048 * HPWP,
      512 * HPWP,
      HPWP, WP, 2560 * HPWP, /*out_pad*/1, /*relu*/1, 512, 512);

  // --- bott: 3x3 2560->512 over concat, BN (no relu), fp32 out ---
  wmma_conv_gemm<1, HPWP><<<dim3(NTILES, 4), 256, 0, stream>>>(
      padBig, apBot, sT, tT, outp, nullptr,
      2560 * HPWP,
      PHW, PW, 512 * PHW, 0, 0, 2560, 512);
}
